// ProteinEditDecoder_86603720556779
// MI455X (gfx1250) — matrix operand order repaired and checked by reading, not compiled, not run on hardware
//
#include <hip/hip_runtime.h>
#include <hip/hip_bf16.h>
#include <math.h>

// ---------------- problem constants ----------------
#define Bn     8
#define Nn     1024
#define Dn     256
#define NODES  (Bn * Nn)      // 8192
#define HEADSn 4

// ---------------- WMMA types (CDNA5, wave32) ----------------
typedef _Float16     v16h __attribute__((ext_vector_type(16)));
typedef float        v8f  __attribute__((ext_vector_type(8)));
typedef unsigned int v8u  __attribute__((ext_vector_type(8)));

union Frag { v8u u; v16h h; };

__device__ __forceinline__ unsigned short f2h(float x) {
  union { _Float16 h; unsigned short u; } c;
  c.h = (_Float16)x;
  return c.u;
}
__device__ __forceinline__ unsigned packh2(float lo, float hi) {
  return (unsigned)f2h(lo) | ((unsigned)f2h(hi) << 16);
}

// =====================================================================
// Tiled WMMA GEMM: C[M,Nc] = epilogue(A[M,K] @ W[K,Nc])
//   epilogue (compile-time flags): (+bias[col]) (+rowadd[batch*Nc+col])
//                                  (relu) (*mask[row])
// Block: 256 threads = 8 waves; block tile 128x64; wave tile 32x32 (2x2 WMMA)
// Staging: batched global_load_b128, f16-pair-packed ds_store_b32.
// Note: BM=128 divides Nn=1024, so every block lies in a single batch.
// =====================================================================
#define BM 128
#define BN 64
#define BK 32

template <bool HAS_BIAS, bool HAS_ROWADD, bool DO_RELU, bool HAS_MASK>
__global__ __launch_bounds__(256)
void gemm_f16_wmma(const float* __restrict__ A, const float* __restrict__ W,
                   const float* __restrict__ bias, const float* __restrict__ rowadd,
                   const unsigned char* __restrict__ mask, float* __restrict__ C,
                   int M, int K, int Nc)
{
  __shared__ unsigned short As[BM][BK + 2];   // [m][k], 68B row stride (4B aligned)
  __shared__ unsigned short Bs[BN][BK + 2];   // [n][k]  (transposed for pair reads)

  const int tid  = threadIdx.x;
  const int m0   = blockIdx.y * BM;
  const int n0   = blockIdx.x * BN;
  const int wid  = tid >> 5;
  const int lane = tid & 31;
  const int wm   = wid & 3;     // wave row-block (0..3) of 32
  const int wn   = wid >> 2;    // wave col-block (0..1) of 32
  const int l16  = lane & 15;
  const int half = lane >> 4;

  // staging coordinates (loop-invariant)
  const int arow = tid >> 1;            // A: row 0..127
  const int acol = (tid & 1) * 16;      // A: col 0 / 16 (16 consecutive floats)
  const int bk   = tid >> 3;            // B: k row 0..31
  const int bn   = (tid & 7) * 8;       // B: 8 consecutive n

  v8f acc[2][2] = {};

  for (int k0 = 0; k0 < K; k0 += BK) {
    // ---- stage A tile: 4x float4 per thread, issued together ----
    const float4* asrc = (const float4*)&A[(size_t)(m0 + arow) * K + k0 + acol];
    float4 f0 = asrc[0], f1 = asrc[1], f2 = asrc[2], f3 = asrc[3];
    // ---- stage B tile: 2x float4 per thread ----
    const float4* bsrc = (const float4*)&W[(size_t)(k0 + bk) * Nc + n0 + bn];
    float4 g0 = bsrc[0], g1 = bsrc[1];

    // prefetch next K-tile while this one is converted/consumed
    if (k0 + BK < K) {
      __builtin_prefetch(&A[(size_t)(m0 + arow) * K + (k0 + BK) + acol], 0, 1);
      __builtin_prefetch(&W[(size_t)(k0 + BK + bk) * Nc + n0 + bn], 0, 1);
    }

    unsigned* ad = (unsigned*)&As[arow][acol];   // 4B aligned
    ad[0] = packh2(f0.x, f0.y); ad[1] = packh2(f0.z, f0.w);
    ad[2] = packh2(f1.x, f1.y); ad[3] = packh2(f1.z, f1.w);
    ad[4] = packh2(f2.x, f2.y); ad[5] = packh2(f2.z, f2.w);
    ad[6] = packh2(f3.x, f3.y); ad[7] = packh2(f3.z, f3.w);

    Bs[bn + 0][bk] = f2h(g0.x); Bs[bn + 1][bk] = f2h(g0.y);
    Bs[bn + 2][bk] = f2h(g0.z); Bs[bn + 3][bk] = f2h(g0.w);
    Bs[bn + 4][bk] = f2h(g1.x); Bs[bn + 5][bk] = f2h(g1.y);
    Bs[bn + 6][bk] = f2h(g1.z); Bs[bn + 7][bk] = f2h(g1.w);
    __syncthreads();

    // Fragments per documented 16-bit layouts:
    //  A 16x32: lane<16 -> M=lane, elems 0..7 = K 0..7, 8..15 = K 16..23;
    //           lane>=16 -> +8 on K base.
    //  B 32x16: same K mapping as A (elems 0..7 = K 0..7, 8..15 = K 16..23; lane>=16 +8); N=lane&15.
    Frag a[2], b[2];
#pragma unroll
    for (int s = 0; s < 2; ++s) {
      const unsigned* ar = (const unsigned*)&As[wm * 32 + s * 16 + l16][0];
#pragma unroll
      for (int p = 0; p < 8; ++p)
        a[s].u[p] = ar[(p & 3) + ((p >> 2) << 3) + half * 4];
      const unsigned* br = (const unsigned*)&Bs[wn * 32 + s * 16 + l16][0];
#pragma unroll
      for (int p = 0; p < 8; ++p)
        b[s].u[p] = br[(p & 3) + ((p >> 2) << 3) + half * 4];
    }

#pragma unroll
    for (int si = 0; si < 2; ++si)
#pragma unroll
      for (int sj = 0; sj < 2; ++sj)
        acc[si][sj] = __builtin_amdgcn_wmma_f32_16x16x32_f16(
            false, a[si].h, false, b[sj].h, (short)0, acc[si][sj], false, false);
    __syncthreads();
  }

  // ---- epilogue (C 16x16 f32 layout: VGPR r -> M = r + half*8, N = lane&15) ----
  const int batch = m0 >> 10;           // whole block lies in one batch

  float coladd[2] = {0.f, 0.f};
#pragma unroll
  for (int sj = 0; sj < 2; ++sj) {
    int gcol = n0 + wn * 32 + sj * 16 + l16;
    if (HAS_BIAS)   coladd[sj] += bias[gcol];
    if (HAS_ROWADD) coladd[sj] += rowadd[batch * Nc + gcol];
  }

#pragma unroll
  for (int si = 0; si < 2; ++si) {
    float mrow[8];
#pragma unroll
    for (int r = 0; r < 8; ++r) {
      int grow = m0 + wm * 32 + si * 16 + r + half * 8;
      mrow[r] = HAS_MASK ? (float)mask[grow] : 1.f;
    }
#pragma unroll
    for (int sj = 0; sj < 2; ++sj) {
      int gcol = n0 + wn * 32 + sj * 16 + l16;
#pragma unroll
      for (int r = 0; r < 8; ++r) {
        int grow = m0 + wm * 32 + si * 16 + r + half * 8;
        float v = acc[si][sj][r] + coladd[sj];
        if (DO_RELU) v = fmaxf(v, 0.f);
        if (HAS_MASK) v *= mrow[r];
        C[(size_t)grow * Nc + gcol] = v;
      }
    }
  }
}

// =====================================================================
// Small per-batch precompute (single block, 256 threads):
//  textf = text_emb@wt+bt ; editp = edit_raw@we+be
//  tvec  = textf@wp[256:512] + editp@wp[512:768] + bp   (folded into node GEMM)
//  attnv = out_proj(v_proj(textf))  -- exact MHA collapse (uniform softmax)
// =====================================================================
__global__ __launch_bounds__(256)
void prep_small(const float* __restrict__ temb, const float* __restrict__ wt, const float* __restrict__ bt,
                const float* __restrict__ eraw, const float* __restrict__ we, const float* __restrict__ be,
                const float* __restrict__ wp,   const float* __restrict__ bp,
                const float* __restrict__ miw,  const float* __restrict__ mib,
                const float* __restrict__ mow,  const float* __restrict__ mob,
                float* textf, float* editp, float* tvec, float* vvec, float* attnv)
{
  int j = threadIdx.x;   // 0..255
  for (int b = 0; b < Bn; ++b) {
    float s = bt[j];
    for (int d = 0; d < 128; ++d) s += temb[b * 128 + d] * wt[d * 256 + j];
    textf[b * 256 + j] = s;
    float e = be[j];
    for (int d = 0; d < 768; ++d) e += eraw[b * 768 + d] * we[d * 256 + j];
    editp[b * 256 + j] = e;
  }
  __syncthreads();
  for (int b = 0; b < Bn; ++b) {
    float s = bp[j];
    for (int d = 0; d < 256; ++d) s += textf[b * 256 + d] * wp[(256 + d) * 256 + j];
    for (int d = 0; d < 256; ++d) s += editp[b * 256 + d] * wp[(512 + d) * 256 + j];
    tvec[b * 256 + j] = s;
    float v = mib[512 + j];
    for (int d = 0; d < 256; ++d) v += textf[b * 256 + d] * miw[(size_t)(512 + j) * 256 + d];
    vvec[b * 256 + j] = v;
  }
  __syncthreads();
  for (int b = 0; b < Bn; ++b) {
    float a = mob[j];
    for (int d = 0; d < 256; ++d) a += vvec[b * 256 + d] * mow[j * 256 + d];
    attnv[b * 256 + j] = a;
  }
}

// xin = node + editp[batch]
__global__ void add_edit(const float* __restrict__ node, const float* __restrict__ editp,
                         float* __restrict__ xin)
{
  int i = blockIdx.x, d = threadIdx.x;
  xin[(size_t)i * 256 + d] = node[(size_t)i * 256 + d] + editp[(i >> 10) * 256 + d];
}

// zero acc, set amax=-inf-ish, den=0
__global__ void init_gat(float* acc, float* amax, float* den)
{
  int t = blockIdx.x * blockDim.x + threadIdx.x;
  int stride = gridDim.x * blockDim.x;
  for (int i = t; i < NODES * 256; i += stride) acc[i] = 0.f;
  if (t < NODES * 4) { amax[t] = -3.0e38f; den[t] = 0.f; }
}

// a_s[i,h] = xw[i,h,:]·att_src[h] ; a_d likewise   (one wave per head)
__global__ __launch_bounds__(128)
void gat_coeff(const float* __restrict__ xw, const float* __restrict__ asv,
               const float* __restrict__ adv, float* __restrict__ a_s, float* __restrict__ a_d)
{
  int i = blockIdx.x;
  int h = threadIdx.x >> 5, lane = threadIdx.x & 31;
  const float* row = xw + (size_t)i * 1024 + h * 256;
  float s = 0.f, d = 0.f;
  for (int j = lane; j < 256; j += 32) {
    float v = row[j];
    s += v * asv[h * 256 + j];
    d += v * adv[h * 256 + j];
  }
  for (int o = 16; o > 0; o >>= 1) {
    s += __shfl_down(s, o, 32);
    d += __shfl_down(d, o, 32);
  }
  if (lane == 0) { a_s[i * 4 + h] = s; a_d[i * 4 + h] = d; }
}

__device__ __forceinline__ void atomicMaxF(float* addr, float v)
{
  int iv = __float_as_int(v);
  if (iv >= 0) atomicMax((int*)addr, iv);
  else         atomicMin((unsigned int*)addr, (unsigned int)iv);
}

__device__ __forceinline__ void edge_sd(const int* ei, int E, int e, int& s, int& d)
{
  if (e < E) { s = ei[e]; d = ei[E + e]; } else { s = d = e - E; }
}

__global__ void edge_max(const int* __restrict__ ei, int E, int Et,
                         const float* __restrict__ a_s, const float* __restrict__ a_d,
                         float* __restrict__ amax)
{
  int e = blockIdx.x * blockDim.x + threadIdx.x;
  if (e >= Et) return;
  int s, d; edge_sd(ei, E, e, s, d);
#pragma unroll
  for (int h = 0; h < 4; ++h) {
    float a = a_s[s * 4 + h] + a_d[d * 4 + h];
    a = a > 0.f ? a : 0.2f * a;          // leaky_relu(0.2)
    atomicMaxF(&amax[d * 4 + h], a);
  }
}

__global__ void edge_exp(const int* __restrict__ ei, int E, int Et,
                         const float* __restrict__ a_s, const float* __restrict__ a_d,
                         const float* __restrict__ amax, float* __restrict__ ew,
                         float* __restrict__ den)
{
  int e = blockIdx.x * blockDim.x + threadIdx.x;
  if (e >= Et) return;
  int s, d; edge_sd(ei, E, e, s, d);
#pragma unroll
  for (int h = 0; h < 4; ++h) {
    float a = a_s[s * 4 + h] + a_d[d * 4 + h];
    a = a > 0.f ? a : 0.2f * a;
    float ex = expf(a - amax[d * 4 + h]);
    ew[(size_t)e * 4 + h] = ex;
    atomicAdd(&den[d * 4 + h], ex);
  }
}

__global__ void edge_norm(const int* __restrict__ ei, int E, int Et,
                          const float* __restrict__ den, float* __restrict__ ew)
{
  int e = blockIdx.x * blockDim.x + threadIdx.x;
  if (e >= Et) return;
  int d = (e < E) ? ei[E + e] : (e - E);
#pragma unroll
  for (int h = 0; h < 4; ++h) ew[(size_t)e * 4 + h] /= den[d * 4 + h];
}

// acc[dst,d] += 0.25 * sum_h w[e,h] * xw[src, h*256+d]   (mean over heads fused)
__global__ __launch_bounds__(256)
void edge_agg(const int* __restrict__ ei, int E, int Et,
              const float* __restrict__ ew, const float* __restrict__ xw,
              float* __restrict__ acc)
{
  int e = blockIdx.x;
  __shared__ int ss, sd;
  __shared__ float w[4];
  if (threadIdx.x == 0) {
    int s, d; edge_sd(ei, E, e, s, d);
    ss = s; sd = d;
  }
  if (threadIdx.x < 4) w[threadIdx.x] = 0.25f * ew[(size_t)e * 4 + threadIdx.x];
  __syncthreads();
  int d0 = threadIdx.x;
  const float* xr = xw + (size_t)ss * 1024;
  float v = w[0] * xr[d0] + w[1] * xr[256 + d0] + w[2] * xr[512 + d0] + w[3] * xr[768 + d0];
  atomicAdd(&acc[(size_t)sd * 256 + d0], v);
}

// x = relu(acc + g_b); optionally xin2 = x + editp[batch]
__global__ void gat_post(const float* __restrict__ acc, const float* __restrict__ gb,
                         const float* __restrict__ editp, float* __restrict__ xout,
                         float* __restrict__ xin2, int make_xin)
{
  int i = blockIdx.x, d = threadIdx.x;
  float v = fmaxf(acc[(size_t)i * 256 + d] + gb[d], 0.f);
  xout[(size_t)i * 256 + d] = v;
  if (make_xin) xin2[(size_t)i * 256 + d] = v + editp[(i >> 10) * 256 + d];
}

// comb = node2 + softmax2(node2·textf, node2·editp)·[textf,editp] + attnv[batch]
__global__ __launch_bounds__(256)
void final_combine(const float* __restrict__ node2, const float* __restrict__ textf,
                   const float* __restrict__ editp, const float* __restrict__ attnv,
                   float* __restrict__ comb)
{
  __shared__ float r0[256], r1[256];
  int i = blockIdx.x, t = threadIdx.x, b = i >> 10;
  float nv = node2[(size_t)i * 256 + t];
  float tf = textf[b * 256 + t], ep = editp[b * 256 + t];
  r0[t] = nv * tf; r1[t] = nv * ep;
  __syncthreads();
  for (int st = 128; st > 0; st >>= 1) {
    if (t < st) { r0[t] += r0[t + st]; r1[t] += r1[t + st]; }
    __syncthreads();
  }
  float s0 = r0[0], s1 = r1[0];
  float m = fmaxf(s0, s1);
  float e0 = expf(s0 - m), e1 = expf(s1 - m);
  float inv = 1.f / (e0 + e1);
  comb[(size_t)i * 256 + t] = nv + (e0 * inv) * tf + (e1 * inv) * ep + attnv[b * 256 + t];
}

__global__ void coords_head(const float* __restrict__ h, const float* __restrict__ cw2,
                            const float* __restrict__ cb2, const unsigned char* __restrict__ mask,
                            float* __restrict__ out)
{
  int tid = blockIdx.x * blockDim.x + threadIdx.x;
  if (tid >= NODES * 3) return;
  int i = tid / 3, c = tid % 3;
  const float* hr = h + (size_t)i * 256;
  float s = cb2[c];
  for (int d = 0; d < 256; ++d) s += hr[d] * cw2[d * 3 + c];
  out[tid] = s * (float)mask[i];
}

__global__ void aa_logits_k(const float* __restrict__ h, const float* __restrict__ aw2,
                            const float* __restrict__ ab2, const unsigned char* __restrict__ mask,
                            float* __restrict__ logits)
{
  int tid = blockIdx.x * blockDim.x + threadIdx.x;
  if (tid >= NODES * 20) return;
  int i = tid / 20, c = tid % 20;
  const float* hr = h + (size_t)i * 256;
  float s = ab2[c];
  for (int d = 0; d < 256; ++d) s += hr[d] * aw2[d * 20 + c];
  logits[tid] = s * (float)mask[i];
}

__global__ void aa_soft(const float* __restrict__ logits, const unsigned char* __restrict__ mask,
                        float* __restrict__ out)
{
  int i = blockIdx.x * blockDim.x + threadIdx.x;
  if (i >= NODES) return;
  float m = -3.0e38f;
  for (int c = 0; c < 20; ++c) m = fmaxf(m, logits[i * 20 + c]);
  float e[20], s = 0.f;
  for (int c = 0; c < 20; ++c) { e[c] = expf(logits[i * 20 + c] - m); s += e[c]; }
  float inv = (float)mask[i] / s;
  for (int c = 0; c < 20; ++c) out[i * 20 + c] = e[c] * inv;
}

// =====================================================================
extern "C" void kernel_launch(void* const* d_in, const int* in_sizes, int n_in,
                              void* d_out, int out_size, void* d_ws, size_t ws_size,
                              hipStream_t stream)
{
  const float* semb = (const float*)d_in[0];   // [8,1024,768]
  const float* temb = (const float*)d_in[1];   // [8,128]
  const float* eraw = (const float*)d_in[3];   // [8,768]
  const float* ws_  = (const float*)d_in[4];
  const float* bs_  = (const float*)d_in[5];
  const float* wt_  = (const float*)d_in[6];
  const float* bt_  = (const float*)d_in[7];
  const float* we_  = (const float*)d_in[8];
  const float* be_  = (const float*)d_in[9];
  const float* wp_  = (const float*)d_in[10];
  const float* bp_  = (const float*)d_in[11];
  const float* g1l  = (const float*)d_in[12];
  const float* g1as = (const float*)d_in[13];
  const float* g1ad = (const float*)d_in[14];
  const float* g1b  = (const float*)d_in[15];
  const float* g2l  = (const float*)d_in[16];
  const float* g2as = (const float*)d_in[17];
  const float* g2ad = (const float*)d_in[18];
  const float* g2b  = (const float*)d_in[19];
  const float* miw  = (const float*)d_in[20];
  const float* mib  = (const float*)d_in[21];
  const float* mow  = (const float*)d_in[22];
  const float* mob  = (const float*)d_in[23];
  const float* cw1  = (const float*)d_in[24];
  const float* cb1  = (const float*)d_in[25];
  const float* cw2  = (const float*)d_in[26];
  const float* cb2  = (const float*)d_in[27];
  const float* aw1  = (const float*)d_in[28];
  const float* ab1  = (const float*)d_in[29];
  const float* aw2  = (const float*)d_in[30];
  const float* ab2  = (const float*)d_in[31];
  const int*   ei   = (const int*)d_in[32];    // [2,E]
  const unsigned char* mask = (const unsigned char*)d_in[33];

  const int E  = in_sizes[32] / 2;
  const int Et = E + NODES;

  // ---- scratch layout (floats), ~70.6 MB ----
  float* w = (float*)d_ws;
  float* bufT   = w; w += (size_t)NODES * 256;   // struct -> xin1 -> xin2 -> mlp hidden
  float* bufN   = w; w += (size_t)NODES * 256;   // node0 -> node2
  float* bufX1  = w; w += (size_t)NODES * 256;   // gat1 out -> comb
  float* bufXW  = w; w += (size_t)NODES * 1024;  // per-head GAT features
  float* bufACC = w; w += (size_t)NODES * 256;   // head-mean aggregation
  float* textf  = w; w += Bn * 256;
  float* editp  = w; w += Bn * 256;
  float* tvec   = w; w += Bn * 256;
  float* vvec   = w; w += Bn * 256;
  float* attnv  = w; w += Bn * 256;
  float* a_s    = w; w += NODES * 4;
  float* a_d    = w; w += NODES * 4;
  float* amax   = w; w += NODES * 4;
  float* den    = w; w += NODES * 4;
  float* ew_    = w; w += (size_t)Et * 4;
  float* logits = w; w += NODES * 20;
  float* outv   = (float*)d_out;

  const dim3 blk(256);
  const int eb = (Et + 255) / 256;

  // per-batch small projections + MHA collapse + folded tvec
  prep_small<<<1, 256, 0, stream>>>(temb, wt_, bt_, eraw, we_, be_, wp_, bp_,
                                    miw, mib, mow, mob, textf, editp, tvec, vvec, attnv);

  // struct = S @ ws + bs
  gemm_f16_wmma<true, false, false, false><<<dim3(256 / BN, NODES / BM), blk, 0, stream>>>(
      semb, ws_, bs_, nullptr, nullptr, bufT, NODES, 768, 256);
  // node = struct @ wp[:256] + tvec[b], masked
  gemm_f16_wmma<false, true, false, true><<<dim3(256 / BN, NODES / BM), blk, 0, stream>>>(
      bufT, wp_, nullptr, tvec, mask, bufN, NODES, 256, 256);
  // xin1 = node + editp[b]
  add_edit<<<NODES, 256, 0, stream>>>(bufN, editp, bufT);

  // ---------------- GAT layer 1 ----------------
  init_gat<<<2048, 256, 0, stream>>>(bufACC, amax, den);
  gemm_f16_wmma<false, false, false, false><<<dim3(1024 / BN, NODES / BM), blk, 0, stream>>>(
      bufT, g1l, nullptr, nullptr, nullptr, bufXW, NODES, 256, 1024);
  gat_coeff<<<NODES, 128, 0, stream>>>(bufXW, g1as, g1ad, a_s, a_d);
  edge_max <<<eb, 256, 0, stream>>>(ei, E, Et, a_s, a_d, amax);
  edge_exp <<<eb, 256, 0, stream>>>(ei, E, Et, a_s, a_d, amax, ew_, den);
  edge_norm<<<eb, 256, 0, stream>>>(ei, E, Et, den, ew_);
  edge_agg <<<Et, 256, 0, stream>>>(ei, E, Et, ew_, bufXW, bufACC);
  gat_post <<<NODES, 256, 0, stream>>>(bufACC, g1b, editp, bufX1, bufT, 1);

  // ---------------- GAT layer 2 ----------------
  init_gat<<<2048, 256, 0, stream>>>(bufACC, amax, den);
  gemm_f16_wmma<false, false, false, false><<<dim3(1024 / BN, NODES / BM), blk, 0, stream>>>(
      bufT, g2l, nullptr, nullptr, nullptr, bufXW, NODES, 256, 1024);
  gat_coeff<<<NODES, 128, 0, stream>>>(bufXW, g2as, g2ad, a_s, a_d);
  edge_max <<<eb, 256, 0, stream>>>(ei, E, Et, a_s, a_d, amax);
  edge_exp <<<eb, 256, 0, stream>>>(ei, E, Et, a_s, a_d, amax, ew_, den);
  edge_norm<<<eb, 256, 0, stream>>>(ei, E, Et, den, ew_);
  edge_agg <<<Et, 256, 0, stream>>>(ei, E, Et, ew_, bufXW, bufACC);
  gat_post <<<NODES, 256, 0, stream>>>(bufACC, g2b, editp, bufN, nullptr, 0);

  // comb = node2 + global-attn(2 ctx tokens) + collapsed-MHA constant
  final_combine<<<NODES, 256, 0, stream>>>(bufN, textf, editp, attnv, bufX1);

  // coords head
  gemm_f16_wmma<true, false, true, false><<<dim3(256 / BN, NODES / BM), blk, 0, stream>>>(
      bufX1, cw1, cb1, nullptr, nullptr, bufT, NODES, 256, 256);
  coords_head<<<(NODES * 3 + 255) / 256, 256, 0, stream>>>(bufT, cw2, cb2, mask, outv);

  // aa head
  gemm_f16_wmma<true, false, true, false><<<dim3(256 / BN, NODES / BM), blk, 0, stream>>>(
      bufX1, aw1, ab1, nullptr, nullptr, bufT, NODES, 256, 256);
  aa_logits_k<<<(NODES * 20 + 255) / 256, 256, 0, stream>>>(bufT, aw2, ab2, mask, logits);
  aa_soft<<<(NODES + 255) / 256, 256, 0, stream>>>(logits, mask, outv + (size_t)NODES * 3);

  (void)n_in; (void)out_size; (void)ws_size;
}
